// Enc_corr_72413148610758
// MI455X (gfx1250) — matrix operand order repaired and checked by reading, not compiled, not run on hardware
//
#include <hip/hip_runtime.h>
#include <hip/hip_bf16.h>

typedef __attribute__((ext_vector_type(16))) _Float16 v16h;
typedef __attribute__((ext_vector_type(8)))  float    v8f;

#define B_    128
#define T_    16
#define S_    10
#define H_    256
#define FPX   64
#define DPX   28
#define CPX   37
#define NPIX  (CPX*CPX)      // 1369 correlation pixels (= K of GEMM1)
#define KPAD  1376           // 43*32 >= 1369
#define KC1   43             // GEMM1 K chunks
#define MTILES_CORR 86       // ceil(1369/16)
#define NROWS (S_*B_*T_)     // 20480 rows (o*T + t)
#define MT1   (NROWS/16)     // 1280 M tiles of GEMM1
// zero-padded LDS frame: 65 rows x 72 halves
#define FROWS 65
#define FSTR  72

// ---------------------------------------------------------------------------
// Prep: W1 -> f16 in WMMA B-fragment order  w1frag[kc][nt][lane][i]
//   k = kc*32 + (i&7) + ((i&8)<<1) + (lane<16 ? 0 : 8)   (B layout K(i,l))
//   n = nt*16 + (lane & 15)
// ---------------------------------------------------------------------------
__global__ void prep_w1(const float* __restrict__ W1, _Float16* __restrict__ w1frag) {
    int idx = blockIdx.x * 256 + threadIdx.x;
    if (idx >= KPAD * H_) return;
    int i  = idx & 15;
    int l  = (idx >> 4) & 31;
    int nt = (idx >> 9) & 15;
    int kc = idx >> 13;
    int k = kc * 32 + (i & 7) + ((i & 8) << 1) + ((l < 16) ? 0 : 8);
    int n = nt * 16 + (l & 15);
    w1frag[idx] = (k < NPIX) ? (_Float16)W1[k * H_ + n] : (_Float16)0.0f;
}

// Zero the K-pad entries (pixels 1369..1375, i.e. kc=42, klocal=25..31) of the
// fragment-ordered conv buffer so GEMM1's last chunk is clean.
__global__ void zero_pad(_Float16* __restrict__ conv) {
    int idx = blockIdx.x * 256 + threadIdx.x;   // MT1 * 16 * 7
    if (idx >= MT1 * 16 * 7) return;
    int o    = idx / 112;
    int rem  = idx - o * 112;
    int lrow = rem / 7;
    int kl   = 25 + (rem - lrow * 7);           // 25..31, kc = 42
    int lh   = (kl >> 3) & 1;                   // A-layout lane half
    int ii   = (kl & 7) | ((kl & 16) >> 1);     // A-layout half index
    size_t a = (((size_t)o * KC1 + 42) * 32 + (lrow + 16 * lh)) * 16 + ii;
    conv[a] = (_Float16)0.0f;
}

// ---------------------------------------------------------------------------
// Stage 1: grouped correlation as per-(group g, t) WMMA GEMM, K chunked by
// template row dy (28 chunks of K=32: dx 0..27 valid, 28..31 zero in B).
// A gather = 16 unconditional ds_load_u16 at constant offsets from one base.
// Output written directly in GEMM1 A-fragment order:
//   conv[mtileM=o][kc][lane = t + 16*lhalf][i],  o = g*S + n, p = pixel = K.
// ---------------------------------------------------------------------------
__global__ __launch_bounds__(256) void corr_wmma(
    const float* __restrict__ frames,       // (B, T, 64, 64)
    const float* __restrict__ digits,       // (S, B, 28, 28) flat (S*B, 784)
    _Float16* __restrict__ conv)            // fragment-ordered, NROWS*KPAD halves
{
    __shared__ __align__(32) _Float16 sF[FROWS * FSTR];   // 9.4 KB, zero padded
    __shared__ __align__(32) _Float16 sB[DPX * 32 * 16];  // 28.7 KB B-fragments

    int blk = blockIdx.x;                 // g*T + t
    int g = blk / T_, t = blk - g * T_;
    int tid = threadIdx.x;

    // frame (g, t) -> LDS f16, zero everywhere outside the 64x64 payload
    const float* fr = frames + ((size_t)g * T_ + t) * (FPX * FPX);
    for (int idx = tid; idx < FROWS * FSTR; idx += 256) {
        int r = idx / FSTR, c = idx - r * FSTR;
        sF[idx] = (r < FPX && c < FPX) ? (_Float16)fr[r * FPX + c] : (_Float16)0.0f;
    }
    // digit templates of output group g, pre-swizzled into B-fragment order:
    // sB[dy][lane][i] = D[dx = (i&7) + ((i&8)<<1) + (lane<16?0:8)][n = lane&15] (0 if dx>=28 or n>=10)
    for (int idx = tid; idx < DPX * 512; idx += 256) {
        int dy  = idx >> 9;
        int rem = idx & 511;
        int l   = rem >> 4;
        int i   = rem & 15;
        int dx  = (i & 7) + ((i & 8) << 1) + ((l < 16) ? 0 : 8);
        int n   = l & 15;
        _Float16 v = (_Float16)0.0f;
        if (dx < DPX && n < S_)
            v = (_Float16)digits[(size_t)(g * S_ + n) * (DPX * DPX) + dy * DPX + dx];
        sB[idx] = v;
    }
    __syncthreads();

    int lane = tid & 31;
    int wave = tid >> 5;
    int hi   = (lane < 16) ? 0 : 1;
    int mrow = lane & 15;

    for (int mt = wave; mt < MTILES_CORR; mt += 8) {
        int p = mt * 16 + mrow;                    // A row = output pixel
        int y = p / CPX, x = p - y * CPX;          // y<=37 even for pad rows
        v8f acc = {0.f, 0.f, 0.f, 0.f, 0.f, 0.f, 0.f, 0.f};
        const _Float16* fp = &sF[y * FSTR + x + (hi ? 8 : 0)];
        for (int dy = 0; dy < DPX; ++dy) {
            v16h a;
#pragma unroll
            for (int i = 0; i < 16; ++i)
                a[i] = fp[i + ((i < 8) ? 0 : 8)];  // constant immediate offsets
            fp += FSTR;
            v16h bfrag = *(const v16h*)&sB[(dy * 32 + lane) * 16];
            acc = __builtin_amdgcn_wmma_f32_16x16x32_f16(
                false, a, false, bfrag, (short)0, acc, false, false);
        }
        // store D into GEMM1 A-fragment order
#pragma unroll
        for (int r = 0; r < 8; ++r) {
            int M  = r + (hi ? 8 : 0);
            int N  = lane & 15;
            int pp = mt * 16 + M;
            if (pp < NPIX && N < S_) {
                int o  = g * S_ + N;                   // = s*B + b
                int kc = pp >> 5, kl = pp & 31;
                int lh = (kl >> 3) & 1;
                int ii = (kl & 7) | ((kl & 16) >> 1);
                size_t aix = (((size_t)o * KC1 + kc) * 32 + (t + 16 * lh)) * 16 + ii;
                conv[aix] = (_Float16)acc[r];
            }
        }
    }
}

// ---------------------------------------------------------------------------
// Stage 2: hidden = tanh(conv @ W1 + b1), all operands pre-swizzled into
// fragment order -> pure aligned 32B global loads, no LDS, no barriers.
// One wave per M tile, all 16 N tiles held in 128 accumulator VGPRs.
// ---------------------------------------------------------------------------
__global__ __launch_bounds__(256) void gemm1_wmma(
    const _Float16* __restrict__ conv,     // [MT1][43][32][16] A-fragments
    const _Float16* __restrict__ w1frag,   // [43][16][32][16]  B-fragments
    const float* __restrict__ b1,
    _Float16* __restrict__ hidden)         // (NROWS, 256) f16
{
    int lane = threadIdx.x & 31;
    int wave = threadIdx.x >> 5;
    int mtile = blockIdx.x * 8 + wave;     // 0..1279

    v8f acc[16];
#pragma unroll
    for (int j = 0; j < 16; ++j)
        acc[j] = (v8f){0.f, 0.f, 0.f, 0.f, 0.f, 0.f, 0.f, 0.f};

    const _Float16* abase = conv + (size_t)mtile * KC1 * 512 + lane * 16;
    for (int kc = 0; kc < KC1; ++kc) {
        v16h afrag = *(const v16h*)(abase + (size_t)kc * 512);
        const _Float16* bbase = w1frag + (size_t)kc * 16 * 512 + lane * 16;
#pragma unroll
        for (int j = 0; j < 16; ++j) {
            v16h bfrag = *(const v16h*)(bbase + j * 512);
            acc[j] = __builtin_amdgcn_wmma_f32_16x16x32_f16(
                false, afrag, false, bfrag, (short)0, acc[j], false, false);
        }
    }

    int hi   = (lane < 16) ? 0 : 1;
    int row0 = mtile * 16 + (hi ? 8 : 0);
#pragma unroll
    for (int j = 0; j < 16; ++j) {
        int h = j * 16 + (lane & 15);
        float bb = b1[h];
#pragma unroll
        for (int r = 0; r < 8; ++r)
            hidden[(size_t)(row0 + r) * H_ + h] = (_Float16)tanhf(acc[j][r] + bb);
    }
}

// ---------------------------------------------------------------------------
// Stage 3: q = tanh(hidden @ W2 + b2); z = q + std * eps with eps replicating
// jax.random.normal(jax.random.key(1)): threefry2x32 key (0,1), split counter,
// u in [nextafter(-1,0), 1), eps = sqrt(2) * erfinv(u).
// ---------------------------------------------------------------------------
__device__ __forceinline__ unsigned rotl32(unsigned x, int r) {
    return (x << r) | (x >> (32 - r));
}

__device__ __forceinline__ void threefry2x32(unsigned k0, unsigned k1,
                                             unsigned x0, unsigned x1,
                                             unsigned* y0, unsigned* y1) {
    unsigned k2 = k0 ^ k1 ^ 0x1BD11BDAu;
    const int RA[4] = {13, 15, 26, 6};
    const int RB[4] = {17, 29, 16, 24};
    x0 += k0; x1 += k1;
#pragma unroll
    for (int i = 0; i < 4; ++i) { x0 += x1; x1 = rotl32(x1, RA[i]); x1 ^= x0; }
    x0 += k1; x1 += k2 + 1u;
#pragma unroll
    for (int i = 0; i < 4; ++i) { x0 += x1; x1 = rotl32(x1, RB[i]); x1 ^= x0; }
    x0 += k2; x1 += k0 + 2u;
#pragma unroll
    for (int i = 0; i < 4; ++i) { x0 += x1; x1 = rotl32(x1, RA[i]); x1 ^= x0; }
    x0 += k0; x1 += k1 + 3u;
#pragma unroll
    for (int i = 0; i < 4; ++i) { x0 += x1; x1 = rotl32(x1, RB[i]); x1 ^= x0; }
    x0 += k1; x1 += k2 + 4u;
#pragma unroll
    for (int i = 0; i < 4; ++i) { x0 += x1; x1 = rotl32(x1, RA[i]); x1 ^= x0; }
    x0 += k2; x1 += k0 + 5u;
    *y0 = x0; *y1 = x1;
}

__device__ __forceinline__ float jax_normal_eps(int j) {
    const int n2 = NROWS;                 // 40960 total / 2
    unsigned i = (unsigned)(j % n2);
    unsigned y0, y1;
    threefry2x32(0u, 1u, i, i + (unsigned)n2, &y0, &y1);
    unsigned bits = (j < n2) ? y0 : y1;
    unsigned f = (bits >> 9) | 0x3f800000u;
    float u01 = __uint_as_float(f) - 1.0f;          // [0,1)
    const float lo = -0.99999994039535522461f;      // nextafter(-1,0)
    float u = u01 * (1.0f - lo) + lo;               // [lo, 1)
    return 1.4142135623730951f * erfinvf(u);
}

__global__ __launch_bounds__(256) void head_kernel(
    const _Float16* __restrict__ hidden,
    const float* __restrict__ W2,
    const float* __restrict__ b2,
    const float* __restrict__ where_std,
    float* __restrict__ out)
{
    int r = blockIdx.x * 256 + threadIdx.x;
    if (r >= NROWS) return;
    const _Float16* hrow = hidden + (size_t)r * H_;
    float q0 = b2[0], q1 = b2[1];
#pragma unroll 4
    for (int k = 0; k < H_; ++k) {
        float h = (float)hrow[k];
        q0 += h * W2[k * 2 + 0];
        q1 += h * W2[k * 2 + 1];
    }
    q0 = tanhf(q0);
    q1 = tanhf(q1);
    out[2 * r + 0] = q0;
    out[2 * r + 1] = q1;
    float e0 = jax_normal_eps(2 * r + 0);
    float e1 = jax_normal_eps(2 * r + 1);
    out[2 * NROWS + 2 * r + 0] = q0 + where_std[0] * e0;
    out[2 * NROWS + 2 * r + 1] = q1 + where_std[1] * e1;
}

// ---------------------------------------------------------------------------
extern "C" void kernel_launch(void* const* d_in, const int* in_sizes, int n_in,
                              void* d_out, int out_size, void* d_ws, size_t ws_size,
                              hipStream_t stream) {
    const float* frames = (const float*)d_in[0];   // (128,16,64,64)
    const float* digits = (const float*)d_in[1];   // (10,128,28,28)
    // d_in[2] = S (scalar, == 10, compiled in)
    const float* W1   = (const float*)d_in[3];     // (1369,256)
    const float* b1   = (const float*)d_in[4];     // (256,)
    const float* W2   = (const float*)d_in[5];     // (256,2)
    const float* b2   = (const float*)d_in[6];     // (2,)
    const float* wstd = (const float*)d_in[7];     // (2,)
    float* out = (float*)d_out;                    // 2*40960 floats

    char* ws = (char*)d_ws;
    size_t conv_bytes = (size_t)NROWS * KPAD * sizeof(_Float16);     // ~56.4 MB
    size_t w1h_bytes  = (size_t)KPAD * H_ * sizeof(_Float16);        // ~0.7 MB
    _Float16* conv   = (_Float16*)ws;
    _Float16* w1frag = (_Float16*)(ws + conv_bytes);
    _Float16* hidden = (_Float16*)(ws + conv_bytes + w1h_bytes);     // ~10.5 MB

    prep_w1<<<(KPAD * H_ + 255) / 256, 256, 0, stream>>>(W1, w1frag);
    zero_pad<<<(MT1 * 16 * 7 + 255) / 256, 256, 0, stream>>>(conv);
    corr_wmma<<<B_ * T_, 256, 0, stream>>>(frames, digits, conv);
    gemm1_wmma<<<MT1 / 8, 256, 0, stream>>>(conv, w1frag, b1, hidden);
    head_kernel<<<(NROWS + 255) / 256, 256, 0, stream>>>(hidden, W2, b2, wstd, out);
}
